// RomHop_42949672961087
// MI455X (gfx1250) — matrix operand order repaired and checked by reading, not compiled, not run on hardware
//
#include <hip/hip_runtime.h>
#include <hip/hip_bf16.h>

typedef __attribute__((ext_vector_type(16))) __bf16   v16bf;
typedef __attribute__((ext_vector_type(8)))  float    v8f;
typedef __attribute__((ext_vector_type(4)))  unsigned v4u;
typedef __attribute__((ext_vector_type(8)))  int      v8i;
typedef __attribute__((ext_vector_type(4)))  int      v4i;

#define B_ 8
#define Q_ 64
#define M_ 4096
#define F_ 8
#define H_ 128
#define NEG_INF (-1e20f)

__device__ __forceinline__ __bf16 f2bf(float f) {
  union { float f; unsigned u; } x; x.f = f;
  unsigned r = (x.u + 0x7FFFu + ((x.u >> 16) & 1u)) >> 16;  // RNE truncate
  unsigned short s = (unsigned short)r;
  __bf16 o; __builtin_memcpy(&o, &s, sizeof(o));
  return o;
}

// LDS byte offset of a __shared__ object (generic AS keeps LDS offset in low 32b)
__device__ __forceinline__ unsigned lds_off(const void* p) {
  return (unsigned)(unsigned long long)p;
}

// ---------------------------------------------------------------------------
// Tensor Data Mover: DMA `nelem` contiguous f32 from global -> LDS.
// D# per CDNA5 ISA: group0 = {count=1, lds_addr,
// global_addr, type=2}; group1 = {data_size=4B, tensor_dim0=nelem,
// tensor_dim1=1, tile_dim0=nelem, tile_dim1=1, dim0_stride=nelem}.
// Per-wave op (EXEC ignored); tracked by TENSORcnt.
// This toolchain exposes the 6-operand builtin (extra int32x8 group).
// ---------------------------------------------------------------------------
__device__ __forceinline__ void tdm_load_f32_1d(unsigned lds_addr,
                                                const float* gptr,
                                                unsigned nelem) {
  unsigned long long ga = (unsigned long long)gptr;
  v4u g0;
  g0[0] = 1u;                                        // count=1, user descriptor
  g0[1] = lds_addr;                                  // LDS byte address
  g0[2] = (unsigned)(ga & 0xFFFFFFFFu);              // global_addr[31:0]
  g0[3] = (unsigned)((ga >> 32) & 0x1FFFFFFu)        // global_addr[56:32]
        | (2u << 30);                                // type=2 ("image")
  v8i g1;
  g1[0] = (int)(2u << 16);                           // data_size=2 -> 4 bytes
  g1[1] = (int)((nelem & 0xFFFFu) << 16);            // tensor_dim0[15:0] @63:48
  g1[2] = (int)(((nelem >> 16) & 0xFFFFu) | (1u << 16)); // dim0[31:16], tensor_dim1=1
  g1[3] = (int)((nelem & 0xFFFFu) << 16);            // tile_dim0 @127:112
  g1[4] = 1;                                         // tile_dim1=1, tile_dim2=0
  g1[5] = (int)nelem;                                // tensor_dim0_stride lo
  g1[6] = 0;
  g1[7] = 0;
  v4i z4 = {0, 0, 0, 0};
  v8i z8 = {0, 0, 0, 0, 0, 0, 0, 0};
  __builtin_amdgcn_tensor_load_to_lds(g0, g1, z4, z4, z8, 0);
}

// ---------------------------------------------------------------------------
// Kernel A: logits via bf16 WMMA + masking + both max reductions + fused
// kb_feature_att softmax -> in_mem base, and out_memory F-sum -> out_mem.
// One block = one batch b and 4 m's (32 logit columns), all 64 q rows.
// Both 16KB memory tiles arrive via TDM (tensor_load_to_lds).
// ---------------------------------------------------------------------------
__global__ __launch_bounds__(256)
void att_reduce_kernel(const float* __restrict__ query,      // [B,Q,H]
                       const float* __restrict__ in_mem,     // [B,M,F,H]
                       const float* __restrict__ out_mem,    // [B,M,F,H]
                       const float* __restrict__ atten_mask, // [B,M]
                       const float* __restrict__ ctx_mask,   // [B,M]
                       const float* __restrict__ query_mask, // [B,Q]
                       float* __restrict__ att_qm,           // [B,Q,M]
                       float* __restrict__ d_inmem,          // [B,M,H]
                       float* __restrict__ d_outmem)         // [B,M,H]
{
  __shared__ __bf16 Abf[Q_ * H_];        // query, bf16 [q][h]
  __shared__ __bf16 Bbf[4 * F_ * H_];    // memory tile bf16 [col][h]
  __shared__ float  Bf32[4 * F_ * H_];   // TDM target: in_memory tile f32
  __shared__ float  Of32[4 * F_ * H_];   // TDM target: out_memory tile f32
  __shared__ float  part[4][2][16];      // per-(qt,ct) column maxes
  __shared__ float  wgt[32];             // kb_feature_att for 4m x 8f

  const int tid  = threadIdx.x;
  const int b    = blockIdx.x >> 10;     // 1024 blocks per batch
  const int mblk = blockIdx.x & 1023;
  const int m0   = mblk * 4;

  // --- kick off both tile DMAs first (completion order: Bf32 then Of32)
  if (tid == 0) {
    tdm_load_f32_1d(lds_off(Bf32), in_mem  + (size_t)(b * M_ + m0) * (F_ * H_),
                    4 * F_ * H_);
    tdm_load_f32_1d(lds_off(Of32), out_mem + (size_t)(b * M_ + m0) * (F_ * H_),
                    4 * F_ * H_);
  }

  // --- stage query (L2-resident) while the DMAs stream
  const float* qp = query + (size_t)b * (Q_ * H_);
  for (int i = tid; i < Q_ * H_; i += 256) Abf[i] = f2bf(qp[i]);

  if ((tid >> 5) == 0) __builtin_amdgcn_s_wait_tensorcnt(1);  // Bf32 landed
  __syncthreads();

  // --- bf16 copy of the in_memory tile for the WMMA B operand
  for (int i = tid; i < 4 * F_ * H_; i += 256) Bbf[i] = f2bf(Bf32[i]);
  __syncthreads();

  const int wave = tid >> 5;
  const int lane = tid & 31;
  const int qt   = wave & 3;             // q-tile 0..3
  const int ct   = wave >> 2;            // col-tile 0..1
  const int half = lane >> 4;
  const int ln   = lane & 15;

  v8f acc = {};
  for (int kk = 0; kk < 4; ++kk) {       // K = 128 in chunks of 32
    v16bf a, bm;
#pragma unroll
    for (int e = 0; e < 16; ++e) {
      int v = e >> 1, sel = e & 1;
      int kA = (v < 4 ? 2 * v : 16 + 2 * (v - 4)) + sel + 8 * half;   // A 16x32
      a[e]  = Abf[(qt * 16 + ln) * H_ + kk * 32 + kA];
      int kB = (v < 4 ? 2 * v : 16 + 2 * (v - 4)) + sel + 8 * half;   // B 32x16
      bm[e] = Bbf[(ct * 16 + ln) * H_ + kk * 32 + kB];
    }
    acc = __builtin_amdgcn_wmma_f32_16x16x32_bf16(false, a, false, bm,
                                                  (short)0, acc, false, false);
  }

  // --- masking (any zero mask => exactly -1e20, matching reference algebra)
  const int colb = ct * 16 + ln;         // 0..31 within block
  const int mloc = colb >> 3;
  const int f    = colb & 7;
  const int m    = m0 + mloc;
  const float am = atten_mask[b * M_ + m];
  const float cm = ctx_mask[b * M_ + m];

  float mval[8];
#pragma unroll
  for (int v = 0; v < 8; ++v) {
    int q = qt * 16 + v + 8 * half;
    float qm = query_mask[b * Q_ + q];
    bool dead = (am == 0.f) || (qm == 0.f) || ((f == 7) && (cm == 0.f));
    mval[v] = dead ? NEG_INF : acc[v];
  }

  // --- att_qm = max over f (groups of 8 lanes)
#pragma unroll
  for (int v = 0; v < 8; ++v) {
    float r = mval[v];
    r = fmaxf(r, __shfl_xor(r, 1, 32));
    r = fmaxf(r, __shfl_xor(r, 2, 32));
    r = fmaxf(r, __shfl_xor(r, 4, 32));
    if ((lane & 7) == 0) {
      int q  = qt * 16 + v + 8 * half;
      int mm = m0 + ct * 2 + ((lane >> 3) & 1);
      att_qm[((size_t)(b * Q_ + q)) * M_ + mm] = r;
    }
  }

  // --- per-(m,f) max over q: reduce 8 accum rows, fold halves, stage per qt
  float cmax = mval[0];
#pragma unroll
  for (int v = 1; v < 8; ++v) cmax = fmaxf(cmax, mval[v]);
  cmax = fmaxf(cmax, __shfl_xor(cmax, 16, 32));
  if (lane < 16) part[qt][ct][lane] = cmax;
  __syncthreads();

  // --- kb_feature_att softmax over f (wave 0); also drain the out_mem DMA
  if (tid < 32) {
    int ct2 = tid >> 4, n = tid & 15;
    float mx = fmaxf(fmaxf(part[0][ct2][n], part[1][ct2][n]),
                     fmaxf(part[2][ct2][n], part[3][ct2][n]));
    float g = mx;
    g = fmaxf(g, __shfl_xor(g, 1, 32));
    g = fmaxf(g, __shfl_xor(g, 2, 32));
    g = fmaxf(g, __shfl_xor(g, 4, 32));
    float e = __expf(mx - g);
    float s = e;
    s += __shfl_xor(s, 1, 32);
    s += __shfl_xor(s, 2, 32);
    s += __shfl_xor(s, 4, 32);
    wgt[tid] = e / s;                    // index == col_in_block == m_local*8+f
  }
  if ((tid >> 5) == 0) __builtin_amdgcn_s_wait_tensorcnt(0);  // Of32 landed
  __syncthreads();

  // --- fused epilogues: in_mem base (weighted) + out_mem (plain F-sum)
#pragma unroll
  for (int rep = 0; rep < 2; ++rep) {
    int idx = tid + rep * 256;           // 0..511 : 4m x 128h
    int ml = idx >> 7, h = idx & 127;
    float s1 = 0.f, s2 = 0.f;
#pragma unroll
    for (int ff = 0; ff < F_; ++ff) {
      s1 += wgt[ml * 8 + ff] * Bf32[(ml * 8 + ff) * H_ + h];
      s2 += Of32[(ml * 8 + ff) * H_ + h];
    }
    size_t o = ((size_t)(b * M_ + m0 + ml)) * H_ + h;
    d_inmem[o]  = s1;
    d_outmem[o] = s2;
  }
}

// ---------------------------------------------------------------------------
// S1: per-(b,q) max/sumexp over M (softmax over M stats)
// ---------------------------------------------------------------------------
__global__ __launch_bounds__(256)
void row_stats_kernel(const float* __restrict__ att_qm,
                      float* __restrict__ row_max, float* __restrict__ row_sum)
{
  __shared__ float red[256];
  const int bq = blockIdx.x;
  const float* p = att_qm + (size_t)bq * M_;
  const int tid = threadIdx.x;
  float loc[16];
  float mx = -3.0e38f;
#pragma unroll
  for (int i = 0; i < 16; ++i) { loc[i] = p[tid + i * 256]; mx = fmaxf(mx, loc[i]); }
  red[tid] = mx; __syncthreads();
  for (int s = 128; s > 0; s >>= 1) {
    if (tid < s) red[tid] = fmaxf(red[tid], red[tid + s]);
    __syncthreads();
  }
  float gmx = red[0]; __syncthreads();
  float sum = 0.f;
#pragma unroll
  for (int i = 0; i < 16; ++i) sum += __expf(loc[i] - gmx);
  red[tid] = sum; __syncthreads();
  for (int s = 128; s > 0; s >>= 1) {
    if (tid < s) red[tid] += red[tid + s];
    __syncthreads();
  }
  if (tid == 0) { row_max[bq] = gmx; row_sum[bq] = red[0]; }
}

// ---------------------------------------------------------------------------
// S2: per-(b,m) max/sumexp over Q (softmax over Q stats)
// ---------------------------------------------------------------------------
__global__ __launch_bounds__(256)
void col_stats_kernel(const float* __restrict__ att_qm,
                      float* __restrict__ col_max, float* __restrict__ col_sum)
{
  int idx = blockIdx.x * 256 + threadIdx.x;   // 0..B*M-1
  int b = idx >> 12;
  int m = idx & (M_ - 1);
  const float* p = att_qm + ((size_t)b * Q_) * M_ + m;
  float mx = -3.0e38f;
  for (int q = 0; q < Q_; ++q) mx = fmaxf(mx, p[(size_t)q * M_]);
  float s = 0.f;
  for (int q = 0; q < Q_; ++q) s += __expf(p[(size_t)q * M_] - mx);
  col_max[idx] = mx; col_sum[idx] = s;
}

// ---------------------------------------------------------------------------
// C2: new_query = query + query_att * softmaxM(att_qm) @ out_mem  (bf16 WMMA)
// One block = one b and 64 h columns; K = M streamed via LDS in chunks of 32.
// ---------------------------------------------------------------------------
__global__ __launch_bounds__(256)
void new_query_kernel(const float* __restrict__ query,
                      const float* __restrict__ query_att,
                      const float* __restrict__ att_qm,
                      const float* __restrict__ row_max,
                      const float* __restrict__ row_sum,
                      const float* __restrict__ d_outmem,   // [B,M,H]
                      float* __restrict__ d_nq)             // [B,Q,H]
{
  __shared__ __bf16 Pbf[Q_ * 32];   // probs chunk [q][32k]
  __shared__ __bf16 Obf[32 * 64];   // out_mem chunk [32k][64h]
  __shared__ float  rmx[Q_], rinv[Q_];

  const int tid = threadIdx.x;
  const int b   = blockIdx.x >> 1;
  const int h0  = (blockIdx.x & 1) * 64;

  if (tid < Q_) {
    rmx[tid]  = row_max[b * Q_ + tid];
    rinv[tid] = 1.f / row_sum[b * Q_ + tid];
  }
  __syncthreads();

  const int wave = tid >> 5, lane = tid & 31;
  const int qt   = wave & 3;
  const int hp   = (wave >> 2) * 2;       // two 16-wide h tiles per wave
  const int half = lane >> 4, ln = lane & 15;

  v8f acc0 = {}, acc1 = {};
  for (int mk = 0; mk < M_; mk += 32) {
    for (int i = tid; i < Q_ * 32; i += 256) {
      int q = i >> 5, k = i & 31;
      float a = att_qm[((size_t)(b * Q_ + q)) * M_ + mk + k];
      Pbf[i] = f2bf(__expf(a - rmx[q]) * rinv[q]);
    }
    for (int i = tid; i < 32 * 64; i += 256) {
      int k = i >> 6, h = i & 63;
      Obf[i] = f2bf(d_outmem[((size_t)(b * M_ + mk + k)) * H_ + h0 + h]);
    }
    __syncthreads();
    v16bf a, b0, b1;
#pragma unroll
    for (int e = 0; e < 16; ++e) {
      int v = e >> 1, sel = e & 1;
      int kA = (v < 4 ? 2 * v : 16 + 2 * (v - 4)) + sel + 8 * half;
      a[e]  = Pbf[(qt * 16 + ln) * 32 + kA];
      int kB = (v < 4 ? 2 * v : 16 + 2 * (v - 4)) + sel + 8 * half;
      b0[e] = Obf[kB * 64 + hp * 16 + ln];
      b1[e] = Obf[kB * 64 + (hp + 1) * 16 + ln];
    }
    acc0 = __builtin_amdgcn_wmma_f32_16x16x32_bf16(false, a, false, b0,
                                                   (short)0, acc0, false, false);
    acc1 = __builtin_amdgcn_wmma_f32_16x16x32_bf16(false, a, false, b1,
                                                   (short)0, acc1, false, false);
    __syncthreads();
  }

#pragma unroll
  for (int v = 0; v < 8; ++v) {
    int q = qt * 16 + v + 8 * half;
    float qa = query_att[b * Q_ + q];
    int h1 = h0 + hp * 16 + ln;
    size_t o1 = ((size_t)(b * Q_ + q)) * H_ + h1;
    d_nq[o1] = query[o1] + qa * acc0[v];
    int h2 = h0 + (hp + 1) * 16 + ln;
    size_t o2 = ((size_t)(b * Q_ + q)) * H_ + h2;
    d_nq[o2] = query[o2] + qa * acc1[v];
  }
}

// ---------------------------------------------------------------------------
// D: in_mem += kb_att * (softmaxQ(att_qm)^T @ new_query)   (bf16 WMMA, K=Q=64)
// One block = one b and 64 m rows.
// ---------------------------------------------------------------------------
__global__ __launch_bounds__(256)
void in_mem_update_kernel(const float* __restrict__ query_att,
                          const float* __restrict__ att_qm,
                          const float* __restrict__ row_max,
                          const float* __restrict__ row_sum,
                          const float* __restrict__ col_max,
                          const float* __restrict__ col_sum,
                          const float* __restrict__ d_nq,     // [B,Q,H]
                          float* __restrict__ d_inmem)        // [B,M,H]
{
  __shared__ __bf16 P2bf[64 * 64];    // probs2^T [m_local][q]
  __shared__ __bf16 NQbf[Q_ * H_];    // new_query bf16 [q][h]
  __shared__ float  kb[64];           // kb_att for this m tile

  const int tid = threadIdx.x;
  const int b   = blockIdx.x >> 6;
  const int m0  = (blockIdx.x & 63) * 64;

  for (int i = tid; i < Q_ * H_; i += 256)
    NQbf[i] = f2bf(d_nq[((size_t)b * Q_) * H_ + i]);

  for (int i = tid; i < 64 * 64; i += 256) {
    int ml = i >> 6, q = i & 63;
    int m = m0 + ml;
    float a  = att_qm[((size_t)(b * Q_ + q)) * M_ + m];
    float p2 = __expf(a - col_max[b * M_ + m]) / col_sum[b * M_ + m];
    P2bf[i] = f2bf(p2);
  }

  if (tid < 64) {
    int m = m0 + tid;
    float s = 0.f;
    for (int q = 0; q < Q_; ++q) {
      float a = att_qm[((size_t)(b * Q_ + q)) * M_ + m];
      float p = __expf(a - row_max[b * Q_ + q]) / row_sum[b * Q_ + q];
      s += query_att[b * Q_ + q] * p;
    }
    kb[tid] = s;
  }
  __syncthreads();

  const int wave = tid >> 5, lane = tid & 31;
  const int mt   = wave & 3;
  const int half = lane >> 4, ln = lane & 15;

  for (int j = 0; j < 4; ++j) {
    int ht = (wave >> 2) * 4 + j;
    v8f acc = {};
#pragma unroll
    for (int kk = 0; kk < 2; ++kk) {
      v16bf a, bm;
#pragma unroll
      for (int e = 0; e < 16; ++e) {
        int v = e >> 1, sel = e & 1;
        int kA = (v < 4 ? 2 * v : 16 + 2 * (v - 4)) + sel + 8 * half;
        a[e]  = P2bf[(mt * 16 + ln) * 64 + kk * 32 + kA];
        int kB = (v < 4 ? 2 * v : 16 + 2 * (v - 4)) + sel + 8 * half;
        bm[e] = NQbf[(kk * 32 + kB) * H_ + ht * 16 + ln];
      }
      acc = __builtin_amdgcn_wmma_f32_16x16x32_bf16(false, a, false, bm,
                                                    (short)0, acc, false, false);
    }
#pragma unroll
    for (int v = 0; v < 8; ++v) {
      int ml = mt * 16 + v + 8 * half;
      int h  = ht * 16 + ln;
      size_t o = ((size_t)(b * M_ + m0 + ml)) * H_ + h;
      d_inmem[o] += kb[ml] * acc[v];
    }
  }
}

// ---------------------------------------------------------------------------
extern "C" void kernel_launch(void* const* d_in, const int* in_sizes, int n_in,
                              void* d_out, int out_size, void* d_ws, size_t ws_size,
                              hipStream_t stream) {
  const float* query      = (const float*)d_in[0];
  const float* in_mem     = (const float*)d_in[1];
  const float* out_mem    = (const float*)d_in[2];
  const float* query_att  = (const float*)d_in[3];
  const float* atten_mask = (const float*)d_in[4];
  const float* ctx_mask   = (const float*)d_in[5];
  const float* query_mask = (const float*)d_in[6];

  // d_out: new_query [B,Q,H] | in_mem [B,M,H] | out_mem [B,M,H]
  float* d_nq     = (float*)d_out;
  float* d_inmem  = d_nq + (size_t)B_ * Q_ * H_;
  float* d_outmem = d_inmem + (size_t)B_ * M_ * H_;

  // workspace: att_qm (8 MB) + softmax stats (~0.25 MB)
  float* ws      = (float*)d_ws;
  float* att_qm  = ws;  ws += (size_t)B_ * Q_ * M_;
  float* row_max = ws;  ws += B_ * Q_;
  float* row_sum = ws;  ws += B_ * Q_;
  float* col_max = ws;  ws += B_ * M_;
  float* col_sum = ws;  ws += B_ * M_;

  att_reduce_kernel<<<B_ * (M_ / 4), 256, 0, stream>>>(
      query, in_mem, out_mem, atten_mask, ctx_mask, query_mask,
      att_qm, d_inmem, d_outmem);
  row_stats_kernel<<<B_ * Q_, 256, 0, stream>>>(att_qm, row_max, row_sum);
  col_stats_kernel<<<B_ * M_ / 256, 256, 0, stream>>>(att_qm, col_max, col_sum);
  new_query_kernel<<<B_ * 2, 256, 0, stream>>>(
      query, query_att, att_qm, row_max, row_sum, d_outmem, d_nq);
  in_mem_update_kernel<<<B_ * (M_ / 64), 256, 0, stream>>>(
      query_att, att_qm, row_max, row_sum, col_max, col_sum, d_nq, d_inmem);
}
